// MVINFONCE_55198919688655
// MI455X (gfx1250) — matrix operand order repaired and checked by reading, not compiled, not run on hardware
//
#include <hip/hip_runtime.h>
#include <hip/hip_bf16.h>

typedef __attribute__((ext_vector_type(16))) __bf16 v16bf;
typedef __attribute__((ext_vector_type(8)))  __bf16 v8bf;
typedef __attribute__((ext_vector_type(8)))  float   v8f;
typedef __attribute__((ext_vector_type(4)))  float   v4f;

#define NTOT   16384
#define NREAL  8192
#define DDIM   128
#define TAUINV 2.0f          // 1/tau, tau = 0.5
#define E2     7.3890560989f // exp(2) = diagonal value of exp(sim/tau)

// ---------------- zero init (row_sum accumulators) ----------------
__global__ void mvnce_zero(float* __restrict__ p, int n) {
    int i = blockIdx.x * blockDim.x + threadIdx.x;
    if (i < n) p[i] = 0.0f;
}

// ---------------- L2-normalize rows, f32 -> bf16 ----------------
// one wave per row: 4 f32 per lane, butterfly-reduce sumsq over 32 lanes
__global__ __launch_bounds__(256) void mvnce_normalize(const float* __restrict__ z,
                                                       __bf16* __restrict__ zb) {
    const int wave = (blockIdx.x * blockDim.x + threadIdx.x) >> 5; // row id
    const int lane = threadIdx.x & 31;
    const float* row = z + (size_t)wave * DDIM + lane * 4;
    v4f x = *(const v4f*)row;
    float ss = x[0]*x[0] + x[1]*x[1] + x[2]*x[2] + x[3]*x[3];
    #pragma unroll
    for (int m = 16; m >= 1; m >>= 1) ss += __shfl_xor(ss, m, 32);
    const float inv = rsqrtf(ss);
    __bf16* out = zb + (size_t)wave * DDIM + lane * 4;
    out[0] = (__bf16)(x[0] * inv);
    out[1] = (__bf16)(x[1] * inv);
    out[2] = (__bf16)(x[2] * inv);
    out[3] = (__bf16)(x[3] * inv);
}

// ---------------- positive-pair dots: pd[i] = <zb[i], zb[i+NREAL]> ----------------
__global__ __launch_bounds__(256) void mvnce_posdot(const __bf16* __restrict__ zb,
                                                    float* __restrict__ pd) {
    const int wave = (blockIdx.x * blockDim.x + threadIdx.x) >> 5; // 0..NREAL-1
    const int lane = threadIdx.x & 31;
    const __bf16* r0 = zb + (size_t)wave * DDIM + lane * 4;
    const __bf16* r1 = zb + (size_t)(wave + NREAL) * DDIM + lane * 4;
    float s = 0.0f;
    #pragma unroll
    for (int t = 0; t < 4; ++t) s += (float)r0[t] * (float)r1[t];
    #pragma unroll
    for (int m = 16; m >= 1; m >>= 1) s += __shfl_xor(s, m, 32);
    if (lane == 0) pd[wave] = s;
}

// ---------------- streaming Gram row-sums via bf16 WMMA ----------------
// Block owns 2 row tiles (32 rows). 8 waves stride over 1024 column tiles.
// Per column tile: 4 chained v_wmma_f32_16x16x32_bf16 per row tile (K=128),
// then exp(2*dot) accumulated per C/D element; reduce over N lanes at the end.
__global__ __launch_bounds__(256) void mvnce_gram_rowsum(const __bf16* __restrict__ zb,
                                                         float* __restrict__ rs) {
    const int lane = threadIdx.x & 31;
    const int wave = threadIdx.x >> 5;   // 0..7
    const int l15  = lane & 15;
    const int hsel = lane >> 4;          // 0 or 1
    const int it0  = blockIdx.x * 2;     // first of 2 row tiles

    // A fragments for both row tiles, all 4 K-steps (ISA 16-bit A layout)
    v16bf a[2][4];
    #pragma unroll
    for (int rt = 0; rt < 2; ++rt) {
        const __bf16* rowp = zb + (size_t)((it0 + rt) * 16 + l15) * DDIM + hsel * 8;
        #pragma unroll
        for (int s = 0; s < 4; ++s) {
            v8bf lo = *(const v8bf*)(rowp + 32 * s);
            v8bf hi = *(const v8bf*)(rowp + 32 * s + 16);
            a[rt][s] = __builtin_shufflevector(lo, hi,
                0,1,2,3,4,5,6,7,8,9,10,11,12,13,14,15);
        }
    }

    float acc[2][8];
    #pragma unroll
    for (int rt = 0; rt < 2; ++rt)
        #pragma unroll
        for (int v = 0; v < 8; ++v) acc[rt][v] = 0.0f;

    for (int j = wave; j < NTOT / 16; j += 8) {
        // B fragments (same K order as A: 8-element K chunks at hsel*8 and hsel*8+16)
        const __bf16* colp = zb + (size_t)(j * 16 + l15) * DDIM + hsel * 8;
        v16bf b[4];
        #pragma unroll
        for (int s = 0; s < 4; ++s) {
            v8bf lo = *(const v8bf*)(colp + 32 * s);
            v8bf hi = *(const v8bf*)(colp + 32 * s + 16);
            b[s] = __builtin_shufflevector(lo, hi,
                0,1,2,3,4,5,6,7,8,9,10,11,12,13,14,15);
        }
        #pragma unroll
        for (int rt = 0; rt < 2; ++rt) {
            v8f c = {};
            #pragma unroll
            for (int s = 0; s < 4; ++s)
                c = __builtin_amdgcn_wmma_f32_16x16x32_bf16(
                        false, a[rt][s], false, b[s], (short)0, c, false, false);
            #pragma unroll
            for (int v = 0; v < 8; ++v)
                acc[rt][v] += __expf(TAUINV * c[v]);
        }
    }

    // reduce over N (lanes within each 16-lane half), then per-row atomic add
    #pragma unroll
    for (int rt = 0; rt < 2; ++rt) {
        #pragma unroll
        for (int v = 0; v < 8; ++v) {
            float x = acc[rt][v];
            x += __shfl_xor(x, 1, 32);
            x += __shfl_xor(x, 2, 32);
            x += __shfl_xor(x, 4, 32);
            x += __shfl_xor(x, 8, 32);
            if (l15 == 0)
                atomicAdd(&rs[(it0 + rt) * 16 + v + hsel * 8], x);
        }
    }
}

// ---------------- final loss reduction ----------------
// loss = mean_i [ log(neg_i) - log(2*exp(pd_i/tau)) ]
//      = mean_i [ log(rs[i]+rs[i+NREAL] - 2*e^2) - (ln2 + 2*pd_i) ]
__global__ __launch_bounds__(256) void mvnce_final(const float* __restrict__ pd,
                                                   const float* __restrict__ rs,
                                                   float* __restrict__ out) {
    __shared__ float sm[256];
    float local = 0.0f;
    for (int i = threadIdx.x; i < NREAL; i += 256) {
        float neg    = rs[i] + rs[i + NREAL] - 2.0f * E2;
        float logpos = 0.69314718056f + TAUINV * pd[i];
        local += __logf(neg) - logpos;
    }
    sm[threadIdx.x] = local;
    __syncthreads();
    for (int s = 128; s > 0; s >>= 1) {
        if (threadIdx.x < s) sm[threadIdx.x] += sm[threadIdx.x + s];
        __syncthreads();
    }
    if (threadIdx.x == 0) {
        out[0] = sm[0] / (float)NREAL;
        out[1] = 0.0f;
    }
}

extern "C" void kernel_launch(void* const* d_in, const int* in_sizes, int n_in,
                              void* d_out, int out_size, void* d_ws, size_t ws_size,
                              hipStream_t stream) {
    (void)in_sizes; (void)n_in; (void)out_size; (void)ws_size;
    const float* z = (const float*)d_in[0];
    float* out = (float*)d_out;

    // workspace layout: bf16 normalized z (4 MB) | pos dots (32 KB) | row sums (64 KB)
    char* w = (char*)d_ws;
    __bf16* zb = (__bf16*)w;
    float*  pd = (float*)(w + (size_t)NTOT * DDIM * 2);
    float*  rs = (float*)(w + (size_t)NTOT * DDIM * 2 + (size_t)NREAL * 4);

    mvnce_zero       <<<(NTOT + 255) / 256, 256, 0, stream>>>(rs, NTOT);
    mvnce_normalize  <<<NTOT / 8,  256, 0, stream>>>(z, zb);   // 8 waves/block, 1 row/wave
    mvnce_posdot     <<<NREAL / 8, 256, 0, stream>>>(zb, pd);
    mvnce_gram_rowsum<<<NTOT / 32, 256, 0, stream>>>(zb, rs);  // 2 row tiles/block
    mvnce_final      <<<1, 256, 0, stream>>>(pd, rs, out);
}
